// GCN_DGL_59717225284235
// MI455X (gfx1250) — matrix operand order repaired and checked by reading, not compiled, not run on hardware
//
#include <hip/hip_runtime.h>

// ---------------------------------------------------------------------------
// GCN (DGL GraphConv, norm='both') for MI455X / gfx1250.
// Pipeline: degrees -> norms -> WMMA GEMM1 -> edge scatter-add -> bias+ELU
//           -> WMMA GEMM2 -> edge scatter-add -> bias+mean.
// GEMMs use v_wmma_f32_16x16x32_bf16 with a hi/lo bf16 split of the fp32
// operands (3 WMMAs per k-tile) for near-fp32 accuracy at negligible cost:
// the workload is scatter/bandwidth bound (~1.4 GB vs ~2 GFLOP).
// ---------------------------------------------------------------------------

typedef __bf16 bf16_t;
typedef __attribute__((ext_vector_type(16))) __bf16 v16bf;
typedef __attribute__((ext_vector_type(8)))  float  v8f;

union BFU { unsigned short u; bf16_t b; };
union V16 { v16bf v; unsigned short s[16]; };

__device__ __forceinline__ unsigned short f2bf_bits(float x) {
  unsigned u = __float_as_uint(x);
  unsigned r = u + 0x7FFFu + ((u >> 16) & 1u);   // round-to-nearest-even
  return (unsigned short)(r >> 16);
}
__device__ __forceinline__ float bf_bits2f(unsigned short h) {
  return __uint_as_float(((unsigned)h) << 16);
}
__device__ __forceinline__ float eluf(float x) {
  return x > 0.0f ? x : expm1f(x);
}

// ---------------------------------------------------------------------------
// Degree accumulation: deg_out[src[e]] += 1, deg_in[dst[e]] += 1
// ---------------------------------------------------------------------------
__global__ __launch_bounds__(256) void degree_kernel(
    const int* __restrict__ src, const int* __restrict__ dst,
    float* __restrict__ deg_out, float* __restrict__ deg_in, int E) {
  int e = blockIdx.x * 256 + threadIdx.x;
  if (e < E) {
    unsafeAtomicAdd(deg_out + src[e], 1.0f);
    unsafeAtomicAdd(deg_in  + dst[e], 1.0f);
  }
}

// norms = rsqrt(max(deg, 1))
__global__ __launch_bounds__(256) void norm_kernel(
    const float* __restrict__ deg_out, const float* __restrict__ deg_in,
    float* __restrict__ snorm, float* __restrict__ dnorm, int n) {
  int i = blockIdx.x * 256 + threadIdx.x;
  if (i < n) {
    snorm[i] = rsqrtf(fmaxf(deg_out[i], 1.0f));
    dnorm[i] = rsqrtf(fmaxf(deg_in[i],  1.0f));
  }
}

// ---------------------------------------------------------------------------
// Row-scaled GEMM:  Y[n, NC] = (X[n, KDIM] * rowScale[n]) @ W[KDIM, NC]
// One wave per 16-row block; W pre-swizzled into LDS in exact CDNA5 bf16
// B-fragment layout (per-lane 32B contiguous). fp32 operands are split into
// bf16 hi/lo; 3 WMMAs per 16x16x32 tile give ~fp32 accuracy.
// ---------------------------------------------------------------------------
template<int KDIM, int NC>
__global__ __launch_bounds__(256) void gemm_rowscale_wmma(
    const float* __restrict__ X, const float* __restrict__ Wm,
    const float* __restrict__ rowScale, float* __restrict__ Y, int nrows) {
  constexpr int KSTEPS = KDIM / 32;
  constexpr int CTS    = NC / 16;
  constexpr int WTOT   = KSTEPS * CTS * 512;   // fragment elements

  __shared__ __align__(32) bf16_t lwhi[WTOT];
  __shared__ __align__(32) bf16_t lwlo[WTOT];

  const int tid = threadIdx.x;

  // Pre-swizzle W into B-fragment layout: [ks][ct][lane][j] with
// element j of lane l = W[ks*32 + (l<16?0:8) + j + (j<8?0:8)][ct*16 + (l&15)].
  for (int i = tid; i < WTOT; i += 256) {
    int j    = i & 15;
    int lane = (i >> 4) & 31;
    int ctks = i >> 9;
    int ct   = ctks % CTS;
    int ks   = ctks / CTS;
    int n    = ct * 16 + (lane & 15);
    int k    = ks * 32 + ((lane < 16) ? 0 : 8) + j + ((j < 8) ? 0 : 8);
    float w  = Wm[k * NC + n];
    unsigned short hb = f2bf_bits(w);
    unsigned short lb = f2bf_bits(w - bf_bits2f(hb));
    BFU t;
    t.u = hb; lwhi[i] = t.b;
    t.u = lb; lwlo[i] = t.b;
  }
  __syncthreads();

  const int wave = tid >> 5;
  const int lane = tid & 31;
  const int rb   = blockIdx.x * 8 + wave;          // 16-row block index
  if (rb * 16 >= nrows) return;                    // wave-uniform: EXEC stays all-ones

  const int   m    = lane & 15;
  const int   row  = rb * 16 + m;
  const int   kbA  = (lane < 16) ? 0 : 8;          // A-fragment K sub-block
  const float s    = rowScale[row];
  const float* xr  = X + (size_t)row * KDIM;

  v8f vzero = {};
  v8f acc[CTS];
#pragma unroll
  for (int c = 0; c < CTS; ++c) acc[c] = vzero;

#pragma unroll
  for (int ks = 0; ks < KSTEPS; ++ks) {
    // A fragment: lane row m, K = ks*32 + kbA + {0..7} and +16..+23
    const float4* p0 = (const float4*)(xr + ks * 32 + kbA);
    const float4* p1 = (const float4*)(xr + ks * 32 + kbA + 16);
    float4 x0 = p0[0], x1 = p0[1], x2 = p1[0], x3 = p1[1];
    float f[16] = { x0.x, x0.y, x0.z, x0.w,  x1.x, x1.y, x1.z, x1.w,
                    x2.x, x2.y, x2.z, x2.w,  x3.x, x3.y, x3.z, x3.w };
    V16 ah, al;
#pragma unroll
    for (int j = 0; j < 16; ++j) {
      float xv = f[j] * s;
      unsigned short hb = f2bf_bits(xv);
      ah.s[j] = hb;
      al.s[j] = f2bf_bits(xv - bf_bits2f(hb));
    }

#pragma unroll
    for (int ct = 0; ct < CTS; ++ct) {
      int off = ((ks * CTS + ct) * 32 + lane) * 16;
      v16bf bh = *(const v16bf*)(lwhi + off);
      v16bf bl = *(const v16bf*)(lwlo + off);
      acc[ct] = __builtin_amdgcn_wmma_f32_16x16x32_bf16(
          false, ah.v, false, bh, (short)0, acc[ct], false, false);
      acc[ct] = __builtin_amdgcn_wmma_f32_16x16x32_bf16(
          false, ah.v, false, bl, (short)0, acc[ct], false, false);
      acc[ct] = __builtin_amdgcn_wmma_f32_16x16x32_bf16(
          false, al.v, false, bh, (short)0, acc[ct], false, false);
    }
  }

  // C/D layout: lanes 0-15 VGPR i -> M=i; lanes 16-31 VGPR i -> M=8+i.
  const int mb = (lane < 16) ? 0 : 8;
#pragma unroll
  for (int ct = 0; ct < CTS; ++ct) {
#pragma unroll
    for (int i = 0; i < 8; ++i) {
      Y[(size_t)(rb * 16 + mb + i) * NC + ct * 16 + m] = acc[ct][i];
    }
  }
}

// ---------------------------------------------------------------------------
// Edge scatter-add: agg[dst[e], :] += hw[src[e], :], float4 per thread.
// ---------------------------------------------------------------------------
__global__ __launch_bounds__(256) void scatter_add_f4(
    const float* __restrict__ hw, const int* __restrict__ src,
    const int* __restrict__ dst, float* __restrict__ agg,
    int fdim, int shift, long total) {
  long idx = (long)blockIdx.x * 256 + threadIdx.x;
  if (idx >= total) return;
  int e  = (int)(idx >> shift);
  int f4 = (int)(idx & ((1 << shift) - 1));
  float4 v = ((const float4*)(hw + (size_t)src[e] * fdim))[f4];
  float* d = agg + (size_t)dst[e] * fdim + f4 * 4;
  unsafeAtomicAdd(d + 0, v.x);
  unsafeAtomicAdd(d + 1, v.y);
  unsafeAtomicAdd(d + 2, v.z);
  unsafeAtomicAdd(d + 3, v.w);
}

// out = elu(agg * dnorm + b), float4 per thread over [N, 64]
__global__ __launch_bounds__(256) void bias_elu_f4(
    const float* __restrict__ agg, const float* __restrict__ dnorm,
    const float* __restrict__ b, float* __restrict__ out, int total4) {
  int idx = blockIdx.x * 256 + threadIdx.x;
  if (idx >= total4) return;
  int node = idx >> 4;           // 64 feats / 4
  int c4   = idx & 15;
  float4 a  = ((const float4*)agg)[idx];
  float4 bb = ((const float4*)b)[c4];
  float  dn = dnorm[node];
  float4 r;
  r.x = eluf(a.x * dn + bb.x);
  r.y = eluf(a.y * dn + bb.y);
  r.z = eluf(a.z * dn + bb.z);
  r.w = eluf(a.w * dn + bb.w);
  ((float4*)out)[idx] = r;
}

// logits[n] = mean_c(agg2[n,c]*dnorm[n] + b2[c]) = (dn*sum(agg2) + sum(b2))/32
__global__ __launch_bounds__(256) void final_mean_kernel(
    const float* __restrict__ agg2, const float* __restrict__ dnorm,
    const float* __restrict__ b2, float* __restrict__ out, int n) {
  int i = blockIdx.x * 256 + threadIdx.x;
  if (i >= n) return;
  const float4* p = (const float4*)(agg2 + (size_t)i * 32);
  float s = 0.0f;
#pragma unroll
  for (int j = 0; j < 8; ++j) { float4 v = p[j]; s += v.x + v.y + v.z + v.w; }
  const float4* pb = (const float4*)b2;
  float sb = 0.0f;
#pragma unroll
  for (int j = 0; j < 8; ++j) { float4 v = pb[j]; sb += v.x + v.y + v.z + v.w; }
  out[i] = (dnorm[i] * s + sb) * 0.03125f;
}

// ---------------------------------------------------------------------------
extern "C" void kernel_launch(void* const* d_in, const int* in_sizes, int n_in,
                              void* d_out, int out_size, void* d_ws, size_t ws_size,
                              hipStream_t stream) {
  const float* h  = (const float*)d_in[0];
  const int*   src = (const int*)d_in[1];
  const int*   dst = (const int*)d_in[2];
  const float* W1 = (const float*)d_in[3];
  const float* b1 = (const float*)d_in[4];
  const float* W2 = (const float*)d_in[5];
  const float* b2 = (const float*)d_in[6];
  float*       out = (float*)d_out;

  const int N = in_sizes[0] / 128;
  const int E = in_sizes[1];

  // Workspace layout (floats); bufA/bufB are reused across pipeline stages.
  float* ws      = (float*)d_ws;
  float* deg_out = ws;                          // [N]
  float* deg_in  = ws + (size_t)N;              // [N]
  float* snorm   = ws + 2 * (size_t)N;          // [N]
  float* dnorm   = ws + 3 * (size_t)N;          // [N]
  float* bufA    = ws + 4 * (size_t)N;          // [N,64]: hw1 -> h1 -> agg2
  float* bufB    = bufA + (size_t)N * 64;       // [N,64]: agg1 -> hw2

  hipMemsetAsync(deg_out, 0, (size_t)2 * N * sizeof(float), stream);   // degrees
  hipMemsetAsync(bufB,    0, (size_t)N * 64 * sizeof(float), stream);  // agg1

  degree_kernel<<<(E + 255) / 256, 256, 0, stream>>>(src, dst, deg_out, deg_in, E);
  norm_kernel<<<(N + 255) / 256, 256, 0, stream>>>(deg_out, deg_in, snorm, dnorm, N);

  const int rowBlocks = (N + 15) / 16;          // 6250
  const int gblocks   = (rowBlocks + 7) / 8;    // 8 waves per 256-thread block

  // Layer 1: hw1 = (h * snorm) @ W1   [N,64]
  gemm_rowscale_wmma<128, 64><<<gblocks, 256, 0, stream>>>(h, W1, snorm, bufA, N);

  // agg1[dst] += hw1[src]
  long tot1 = (long)E * 16;
  scatter_add_f4<<<(unsigned)((tot1 + 255) / 256), 256, 0, stream>>>(
      bufA, src, dst, bufB, 64, 4, tot1);

  // h1 = elu(agg1 * dnorm + b1)  -> bufA
  int tot_b = N * 16;
  bias_elu_f4<<<(tot_b + 255) / 256, 256, 0, stream>>>(bufB, dnorm, b1, bufA, tot_b);

  // Layer 2: hw2 = (h1 * snorm) @ W2   [N,32] -> bufB
  gemm_rowscale_wmma<64, 32><<<gblocks, 256, 0, stream>>>(bufA, W2, snorm, bufB, N);

  // agg2 (reuses bufA, safe: h1 fully consumed by GEMM2 in stream order)
  hipMemsetAsync(bufA, 0, (size_t)N * 32 * sizeof(float), stream);
  long tot2 = (long)E * 8;
  scatter_add_f4<<<(unsigned)((tot2 + 255) / 256), 256, 0, stream>>>(
      bufB, src, dst, bufA, 32, 3, tot2);

  // logits = mean over 32 classes
  final_mean_kernel<<<(N + 255) / 256, 256, 0, stream>>>(bufA, dnorm, b2, out, N);
}
